// scGraphformerConv_31533649887564
// MI455X (gfx1250) — matrix operand order repaired and checked by reading, not compiled, not run on hardware
//
#include <hip/hip_runtime.h>
#include <hip/hip_bf16.h>

// ---------------------------------------------------------------------------
// Problem constants (from reference): N=50000, C=256, H=4, D=64, E=800000
// ---------------------------------------------------------------------------

typedef __attribute__((ext_vector_type(16))) __bf16 v16bf;
typedef __attribute__((ext_vector_type(8)))  __bf16 v8bf;
typedef __attribute__((ext_vector_type(8)))  float  v8f;
typedef unsigned int u32;
typedef __attribute__((ext_vector_type(4))) u32 u32x4;
typedef __attribute__((ext_vector_type(8))) u32 u32x8;

__device__ __forceinline__ __bf16 f2bf(float f) {
    unsigned u = __builtin_bit_cast(unsigned, f);
    u += 0x7FFFu + ((u >> 16) & 1u);            // round-to-nearest-even
    unsigned short h = (unsigned short)(u >> 16);
    return __builtin_bit_cast(__bf16, h);
}

__device__ __forceinline__ v8f wmma_bf16(v16bf a, v16bf b, v8f c) {
    // D = A(16x32 bf16) * B(32x16 bf16) + C(16x16 f32)
    return __builtin_amdgcn_wmma_f32_16x16x32_bf16(false, a, false, b,
                                                   (short)0, c, false, false);
}

// A-fragment: 16 contiguous bf16 (K-offsets {kb..kb+15})
__device__ __forceinline__ v16bf load_a_bf(const __bf16* p) {
    v8bf lo = *(const v8bf*)p;
    v8bf hi = *(const v8bf*)(p + 8);
    return __builtin_shufflevector(lo, hi, 0, 1, 2, 3, 4, 5, 6, 7,
                                   8, 9, 10, 11, 12, 13, 14, 15);
}

// ---------------------------------------------------------------------------
// Tensor Data Mover: async 2D tile load (global f32 -> LDS), wave-level op.
// Descriptor per CDNA5 ISA ch.8 (D# group0 = 4 SGPRs, group1 = 8 SGPRs,
// groups 2/3 omitted for <=2D tensors).  Rows beyond tensor_dim1 zero-fill.
// ---------------------------------------------------------------------------
__device__ __forceinline__ void tdm_load_2d_f32(u32 lds_addr, const void* gaddr,
                                                u32 tdim0, u32 tdim1, u32 stride,
                                                u32 tile0, u32 tile1)
{
    unsigned long long ga = (unsigned long long)(uintptr_t)gaddr;
    u32x4 g0;
    g0[0] = 0x1u;                                          // count=1, user desc
    g0[1] = lds_addr;                                      // LDS byte address
    g0[2] = (u32)(ga & 0xFFFFFFFFu);                       // global_addr[31:0]
    g0[3] = (u32)((ga >> 32) & 0x1FFFFFFu) | (2u << 30);   // addr[56:32] | type=2
    u32x8 g1;
    g1[0] = (2u << 16);                                    // data_size = 4 bytes
    g1[1] = (tdim0 & 0xFFFFu) << 16;                       // tensor_dim0[15:0]
    g1[2] = ((tdim0 >> 16) & 0xFFFFu) | ((tdim1 & 0xFFFFu) << 16);
    g1[3] = ((tdim1 >> 16) & 0xFFFFu) | ((tile0 & 0xFFFFu) << 16); // tile_dim0
    g1[4] = (tile1 & 0xFFFFu);                             // tile_dim1, tile_dim2=0
    g1[5] = stride;                                        // dim0_stride[31:0]
    g1[6] = 0;                                             // stride hi, dim1_stride
    g1[7] = 0;
    asm volatile("tensor_load_to_lds %0, %1" :: "s"(g0), "s"(g1) : "memory");
}

// ---------------------------------------------------------------------------
// K0a: weights fp32 -> bf16 (row-major W[o][c] is already the B^T layout).
// ---------------------------------------------------------------------------
__global__ void k_wcvt(const float* __restrict__ wq, const float* __restrict__ wk,
                       const float* __restrict__ wv, __bf16* __restrict__ wbf) {
    int idx = blockIdx.x * 256 + threadIdx.x;          // < 3*65536
    int mat = idx >> 16;
    int i   = idx & 65535;
    const float* src = (mat == 0) ? wq : (mat == 1) ? wk : wv;
    wbf[idx] = f2bf(src[i]);
}

// ---------------------------------------------------------------------------
// K0b: activations fp32 -> bf16 (one pass; removes all per-iteration cvt VALU
// from the GEMM and halves its A-side load bytes).  grid.y: 0=query 1=source.
// ---------------------------------------------------------------------------
__global__ void k_xcvt(const float* __restrict__ xq, const float* __restrict__ xs,
                       __bf16* __restrict__ xqb, __bf16* __restrict__ xsb, int total)
{
    const float* src = blockIdx.y ? xs : xq;
    __bf16* dst = blockIdx.y ? xsb : xqb;
    int base = (blockIdx.x * 256 + threadIdx.x) * 8;
    if (base >= total) return;
    const float4* p = (const float4*)(src + base);
    float4 a = p[0], b = p[1];
    v8bf o;
    o[0] = f2bf(a.x); o[1] = f2bf(a.y); o[2] = f2bf(a.z); o[3] = f2bf(a.w);
    o[4] = f2bf(b.x); o[5] = f2bf(b.y); o[6] = f2bf(b.z); o[7] = f2bf(b.w);
    *(v8bf*)(dst + base) = o;
}

// ---------------------------------------------------------------------------
// K1: QKV GEMM.  out[n,o] = sum_c X[n,c]*W[o,c] + b[o].
// Block: 256 thr = 8 waves; tile M=128 x N=64; wave = 32x32 (2x2 WMMA tiles);
// K loop over 256 in steps of 32.  grid = (ceil(N/128), 4, 3).
// ---------------------------------------------------------------------------
__global__ __launch_bounds__(256) void k_gemm_qkv(
    const __bf16* __restrict__ Xqb, const __bf16* __restrict__ Xsb,
    const float* __restrict__ bq, const float* __restrict__ bk,
    const float* __restrict__ bv, const __bf16* __restrict__ wbf,
    float* __restrict__ q, float* __restrict__ k, float* __restrict__ v, int Nn)
{
    const int z = blockIdx.z;                                   // 0=q 1=k 2=v
    const __bf16* X    = (z == 0) ? Xqb : Xsb;
    const __bf16* W    = wbf + (size_t)z * 65536;
    const float*  bias = (z == 0) ? bq : (z == 1) ? bk : bv;
    float*        out  = (z == 0) ? q  : (z == 1) ? k  : v;

    const int m0   = blockIdx.x * 128;
    const int o0   = blockIdx.y * 64;
    const int w    = threadIdx.x >> 5;
    const int lane = threadIdx.x & 31;
    const int mwave = m0 + (w & 3) * 32;
    const int nwave = o0 + (w >> 2) * 32;
    const int kb_a = (lane < 16) ? 0 : 16;
    const int kb_b = (lane < 16) ? 0 : 16;
    const int arow0 = mwave + (lane & 15);
    const int brow0 = nwave + (lane & 15);

    const int r0 = min(arow0,      Nn - 1);
    const int r1 = min(arow0 + 16, Nn - 1);

    v8f c00 = {}, c01 = {}, c10 = {}, c11 = {};
    for (int k0 = 0; k0 < 256; k0 += 32) {
        v16bf a0 = load_a_bf(X + (size_t)r0 * 256 + k0 + kb_a);
        v16bf a1 = load_a_bf(X + (size_t)r1 * 256 + k0 + kb_a);
        v16bf b0 = *(const v16bf*)(W + (size_t)brow0        * 256 + k0 + kb_b);
        v16bf b1 = *(const v16bf*)(W + (size_t)(brow0 + 16) * 256 + k0 + kb_b);
        c00 = wmma_bf16(a0, b0, c00);
        c01 = wmma_bf16(a0, b1, c01);
        c10 = wmma_bf16(a1, b0, c10);
        c11 = wmma_bf16(a1, b1, c11);
    }

    const int colA = nwave + (lane & 15);
    const int colB = colA + 16;
    const float biasA = bias[colA];
    const float biasB = bias[colB];
    const int rbase0 = mwave + ((lane >> 4) << 3);
    const int rbase1 = rbase0 + 16;
#pragma unroll
    for (int r = 0; r < 8; ++r) {
        int ra = rbase0 + r, rb = rbase1 + r;
        if (ra < Nn) {
            out[(size_t)ra * 256 + colA] = c00[r] + biasA;
            out[(size_t)ra * 256 + colB] = c01[r] + biasB;
        }
        if (rb < Nn) {
            out[(size_t)rb * 256 + colA] = c10[r] + biasA;
            out[(size_t)rb * 256 + colB] = c11[r] + biasB;
        }
    }
}

// ---------------------------------------------------------------------------
// K2: global reductions: sum(q^2), sum(k^2), colsum(k), colsum(v); also emits
// the bf16 copy of q used by the attention WMMA (fused: q is already in regs).
// ---------------------------------------------------------------------------
__global__ __launch_bounds__(256) void k_reduce(
    const float* __restrict__ q, const float* __restrict__ k,
    const float* __restrict__ v, __bf16* __restrict__ qbf,
    float* __restrict__ csk, float* __restrict__ csv,
    float* __restrict__ sums, int Nn)
{
    __shared__ float red[256];
    const int t  = threadIdx.x;
    const int r0 = blockIdx.x * 128;
    float sq = 0.f, sk2 = 0.f, ck = 0.f, cv = 0.f;
    for (int r = 0; r < 128; ++r) {
        int row = r0 + r;
        if (row >= Nn) break;
        float qv = q[(size_t)row * 256 + t];
        float kv = k[(size_t)row * 256 + t];
        float vv = v[(size_t)row * 256 + t];
        qbf[(size_t)row * 256 + t] = f2bf(qv);
        sq += qv * qv; sk2 += kv * kv; ck += kv; cv += vv;
    }
    atomicAdd(&csk[t], ck);
    atomicAdd(&csv[t], cv);
    red[t] = sq; __syncthreads();
    for (int s = 128; s > 0; s >>= 1) { if (t < s) red[t] += red[t + s]; __syncthreads(); }
    if (t == 0) atomicAdd(&sums[0], red[0]);
    __syncthreads();
    red[t] = sk2; __syncthreads();
    for (int s = 128; s > 0; s >>= 1) { if (t < s) red[t] += red[t + s]; __syncthreads(); }
    if (t == 0) atomicAdd(&sums[1], red[0]);
}

// ---------------------------------------------------------------------------
// K3: kvs[h][m][d] = sum_l k[l,h*64+m] * v[l,h*64+d]  (raw, un-normalized).
// TDM stages the 64x64 f32 tiles of k and v into LDS (zero-fill past N),
// each thread accumulates 16 (m,d) products, atomics into global kvs.
// grid = (ceil(N/64), 1, 4).
// ---------------------------------------------------------------------------
__global__ __launch_bounds__(256) void k_kvs(
    const float* __restrict__ k, const float* __restrict__ v,
    float* __restrict__ kvs, int Nn)
{
    __shared__ float smem[2 * 64 * 64];
    float* sk = smem;
    float* sv = smem + 4096;
    const int h  = blockIdx.z;
    const int l0 = blockIdx.x * 64;
    const int t  = threadIdx.x;

    if (t < 32) {                       // wave 0 issues the two DMA descriptors
        u32 sk_off = (u32)(uintptr_t)(void*)sk;
        u32 sv_off = sk_off + 4096u * 4u;
        u32 rows   = (u32)(Nn - l0);    // hardware zero-fills rows beyond this
        const float* gk = k + (size_t)l0 * 256 + h * 64;
        const float* gv = v + (size_t)l0 * 256 + h * 64;
        tdm_load_2d_f32(sk_off, gk, 64u, rows, 256u, 64u, 64u);
        tdm_load_2d_f32(sv_off, gv, 64u, rows, 256u, 64u, 64u);
        __builtin_amdgcn_s_wait_tensorcnt(0);
    }
    __syncthreads();

    const int m  = t & 63;
    const int dg = (t >> 6) * 16;
    float acc[16] = {};
    for (int l = 0; l < 64; ++l) {
        float kk = sk[l * 64 + m];
        const float4* vp = (const float4*)&sv[l * 64 + dg];
        float4 a = vp[0], b = vp[1], c = vp[2], d = vp[3];
        acc[0]  += kk * a.x;  acc[1]  += kk * a.y;  acc[2]  += kk * a.z;  acc[3]  += kk * a.w;
        acc[4]  += kk * b.x;  acc[5]  += kk * b.y;  acc[6]  += kk * b.z;  acc[7]  += kk * b.w;
        acc[8]  += kk * c.x;  acc[9]  += kk * c.y;  acc[10] += kk * c.z;  acc[11] += kk * c.w;
        acc[12] += kk * d.x;  acc[13] += kk * d.y;  acc[14] += kk * d.z;  acc[15] += kk * d.w;
    }
#pragma unroll
    for (int j = 0; j < 16; ++j)
        atomicAdd(&kvs[(size_t)h * 4096 + m * 64 + dg + j], acc[j]);
}

// ---------------------------------------------------------------------------
// K3b: normalizer[n][h] = dot(q[n], colsum_k over head h) / (||q||*||k||) + N
// One wave per node.
// ---------------------------------------------------------------------------
__global__ __launch_bounds__(256) void k_norm(
    const float* __restrict__ q, const float* __restrict__ csk,
    const float* __restrict__ sums, float* __restrict__ normarr, int Nn)
{
    __shared__ float s_csk[256];
    const int t = threadIdx.x;
    s_csk[t] = csk[t];
    __syncthreads();
    const int w = t >> 5, lane = t & 31;
    const int n = blockIdx.x * 8 + w;
    if (n >= Nn) return;
    float acc[4] = {0.f, 0.f, 0.f, 0.f};
#pragma unroll
    for (int j = 0; j < 8; ++j) {
        int c = j * 32 + lane;
        acc[j >> 1] += q[(size_t)n * 256 + c] * s_csk[c];
    }
    const float scale = rsqrtf(sums[0]) * rsqrtf(sums[1]);
#pragma unroll
    for (int h = 0; h < 4; ++h) {
        float x = acc[h];
        for (int off = 16; off > 0; off >>= 1) x += __shfl_xor(x, off, 32);
        if (lane == 0) normarr[(size_t)n * 4 + h] = x * scale + (float)Nn;
    }
}

// ---------------------------------------------------------------------------
// K3c: kvsT_bf[h][d][m] = bf16( kvs[h][m][d] / ||k|| )   (transposed B layout)
// ---------------------------------------------------------------------------
__global__ void k_kvs_cvt(const float* __restrict__ kvs,
                          const float* __restrict__ sums,
                          __bf16* __restrict__ kvsT)
{
    const float inv_kn = rsqrtf(sums[1]);
    for (int idx = blockIdx.x * 256 + threadIdx.x; idx < 4 * 64 * 64;
         idx += gridDim.x * 256) {
        int h = idx >> 12, rem = idx & 4095, d = rem >> 6, m = rem & 63;
        kvsT[idx] = f2bf(kvs[(h << 12) + m * 64 + d] * inv_kn);
    }
}

// ---------------------------------------------------------------------------
// K4: attention: out[n,d] += 0.25 * ( (q@kvs_norm)[n,h,d]/||q|| + vsum[h,d] )
//                              / normalizer[n,h]
// WMMA: A = qbf rows, B = kvsT (bf16, 32KB total -> lives in L2/WGP$).
// Block: 8 waves x 16 rows = 128 rows; wave covers all 4 d-tiles, K=64.
// grid = (ceil(N/128), 1, 4).
// ---------------------------------------------------------------------------
__global__ __launch_bounds__(256) void k_attn(
    const __bf16* __restrict__ qbf, const __bf16* __restrict__ kvsT,
    const float* __restrict__ csv, const float* __restrict__ normarr,
    const float* __restrict__ sums, float* __restrict__ out, int Nn)
{
    const int h    = blockIdx.z;
    const int m0   = blockIdx.x * 128;
    const int w    = threadIdx.x >> 5;
    const int lane = threadIdx.x & 31;
    const int mw   = m0 + w * 16;
    const int kb_a = (lane < 16) ? 0 : 16;
    const int kb_b = (lane < 16) ? 0 : 16;
    const int arow = min(mw + (lane & 15), Nn - 1);
    const __bf16* Bh = kvsT + (size_t)h * 4096;

    v8f c[4] = {{}, {}, {}, {}};
#pragma unroll
    for (int ks = 0; ks < 2; ++ks) {
        v16bf a = load_a_bf(qbf + (size_t)arow * 256 + h * 64 + ks * 32 + kb_a);
#pragma unroll
        for (int nt = 0; nt < 4; ++nt) {
            v16bf b = *(const v16bf*)(Bh + (size_t)(nt * 16 + (lane & 15)) * 64
                                      + ks * 32 + kb_b);
            c[nt] = wmma_bf16(a, b, c[nt]);
        }
    }

    const float inv_qn = rsqrtf(sums[0]);
    const int rbase = mw + ((lane >> 4) << 3);
#pragma unroll
    for (int nt = 0; nt < 4; ++nt) {
        int col = nt * 16 + (lane & 15);
        float vs = csv[h * 64 + col];
#pragma unroll
        for (int r = 0; r < 8; ++r) {
            int row = rbase + r;
            if (row < Nn) {
                float t  = c[nt][r] * inv_qn + vs;
                float nm = normarr[(size_t)row * 4 + h];
                atomicAdd(&out[(size_t)row * 64 + col], 0.25f * t / nm);
            }
        }
    }
}

// ---------------------------------------------------------------------------
// GCN part
// ---------------------------------------------------------------------------
__global__ void k_deg(const int* __restrict__ ei, float* __restrict__ deg, int Ed) {
    int e = blockIdx.x * 256 + threadIdx.x;
    if (e < Ed) atomicAdd(&deg[ei[Ed + e]], 1.0f);   // in-degree on col
}

__global__ void k_vbar(const float* __restrict__ v, float* __restrict__ vbar, int Nn) {
    int idx = blockIdx.x * 256 + threadIdx.x;
    if (idx >= Nn * 64) return;
    int n = idx >> 6, d = idx & 63;
    const float* p = v + (size_t)n * 256 + d;
    vbar[idx] = 0.25f * (p[0] + p[64] + p[128] + p[192]);
}

// one wave per edge: coalesced vbar row read + coalesced atomics on out row
__global__ __launch_bounds__(256) void k_scatter(
    const int* __restrict__ ei, const float* __restrict__ ew,
    const float* __restrict__ deg, const float* __restrict__ vbar,
    float* __restrict__ out, int Ed)
{
    const int w = threadIdx.x >> 5, lane = threadIdx.x & 31;
    const int e = blockIdx.x * 8 + w;
    if (e >= Ed) return;
    const int row = ei[e];
    const int col = ei[Ed + e];
    float val = ew[e] * rsqrtf(deg[row]) * rsqrtf(deg[col]);
    if (!(val == val) || fabsf(val) > 3.0e38f) val = 0.0f;   // nan_to_num
    float a0 = vbar[(size_t)row * 64 + lane]      * val;
    float a1 = vbar[(size_t)row * 64 + lane + 32] * val;
    atomicAdd(&out[(size_t)col * 64 + lane],      a0);
    atomicAdd(&out[(size_t)col * 64 + lane + 32], a1);
}

// ---------------------------------------------------------------------------
// Host-side orchestration
// ---------------------------------------------------------------------------
extern "C" void kernel_launch(void* const* d_in, const int* in_sizes, int n_in,
                              void* d_out, int out_size, void* d_ws, size_t ws_size,
                              hipStream_t stream)
{
    const float* Xq = (const float*)d_in[0];
    const float* Xs = (const float*)d_in[1];
    const float* Wq = (const float*)d_in[2];
    const float* bq = (const float*)d_in[3];
    const float* Wk = (const float*)d_in[4];
    const float* bk = (const float*)d_in[5];
    const float* Wv = (const float*)d_in[6];
    const float* bv = (const float*)d_in[7];
    const int*   ei = (const int*)d_in[8];
    const float* ew = (const float*)d_in[9];
    float* out = (float*)d_out;

    const int Nn = in_sizes[0] / 256;   // 50000
    const int Ed = in_sizes[9];         // 800000

    // workspace carve-out
    char* ws = (char*)d_ws;
    size_t off = 0;
    auto carve = [&](size_t bytes) -> char* {
        off = (off + 255) & ~(size_t)255;
        char* p = ws + off;
        off += bytes;
        return p;
    };
    float*  qf      = (float*)carve((size_t)Nn * 256 * 4);
    float*  kf      = (float*)carve((size_t)Nn * 256 * 4);
    float*  vf      = (float*)carve((size_t)Nn * 256 * 4);
    __bf16* Xqb     = (__bf16*)carve((size_t)Nn * 256 * 2);
    __bf16* Xsb     = (__bf16*)carve((size_t)Nn * 256 * 2);
    __bf16* qbf     = (__bf16*)carve((size_t)Nn * 256 * 2);
    __bf16* wbf     = (__bf16*)carve((size_t)3 * 65536 * 2);
    float*  kvs     = (float*)carve((size_t)4 * 64 * 64 * 4);
    __bf16* kvsT    = (__bf16*)carve((size_t)4 * 64 * 64 * 2);
    float*  csk     = (float*)carve(256 * 4);
    float*  csv     = (float*)carve(256 * 4);
    float*  sums    = (float*)carve(2 * 4);
    float*  normarr = (float*)carve((size_t)Nn * 4 * 4);
    float*  vbar    = (float*)carve((size_t)Nn * 64 * 4);
    float*  deg     = (float*)carve((size_t)Nn * 4);

    // zero the accumulators + output (graph-capture-safe async memsets)
    hipMemsetAsync(kvs,  0, (size_t)4 * 64 * 64 * 4, stream);
    hipMemsetAsync(csk,  0, 256 * 4, stream);
    hipMemsetAsync(csv,  0, 256 * 4, stream);
    hipMemsetAsync(sums, 0, 2 * 4, stream);
    hipMemsetAsync(deg,  0, (size_t)Nn * 4, stream);
    hipMemsetAsync(out,  0, (size_t)Nn * 64 * 4, stream);

    const int mblk = (Nn + 127) / 128;

    // precision conversion prepasses
    k_wcvt<<<(3 * 65536) / 256, 256, 0, stream>>>(Wq, Wk, Wv, wbf);
    k_xcvt<<<dim3((Nn * 256 / 8 + 255) / 256, 2), 256, 0, stream>>>(
        Xq, Xs, Xqb, Xsb, Nn * 256);

    // QKV projection (WMMA, bf16 in / f32 out)
    k_gemm_qkv<<<dim3(mblk, 4, 3), 256, 0, stream>>>(Xqb, Xsb, bq, bk, bv, wbf,
                                                     qf, kf, vf, Nn);
    // global reductions + fused q->bf16
    k_reduce<<<mblk, 256, 0, stream>>>(qf, kf, vf, qbf, csk, csv, sums, Nn);

    // kvs = k^T v per head (TDM-staged LDS tiles)
    k_kvs<<<dim3((Nn + 63) / 64, 1, 4), 256, 0, stream>>>(kf, vf, kvs, Nn);

    // per-node, per-head normalizer
    k_norm<<<(Nn + 7) / 8, 256, 0, stream>>>(qf, csk, sums, normarr, Nn);

    // normalized + transposed bf16 kvs for WMMA-B
    k_kvs_cvt<<<64, 256, 0, stream>>>(kvs, sums, kvsT);

    // attention output (WMMA), accumulated into out
    k_attn<<<dim3(mblk, 1, 4), 256, 0, stream>>>(qbf, kvsT, csv, normarr, sums,
                                                 out, Nn);

    // GCN: degree, head-mean of v, edge scatter
    k_deg<<<(Ed + 255) / 256, 256, 0, stream>>>(ei, deg, Ed);
    k_vbar<<<(Nn * 64 + 255) / 256, 256, 0, stream>>>(vf, vbar, Nn);
    k_scatter<<<(Ed + 7) / 8, 256, 0, stream>>>(ei, ew, deg, vbar, out, Ed);
}
